// CSSN_56264071577834
// MI455X (gfx1250) — matrix operand order repaired and checked by reading, not compiled, not run on hardware
//
#include <hip/hip_runtime.h>
#include <hip/hip_bf16.h>
#include <math.h>

typedef _Float16 v16h __attribute__((ext_vector_type(16)));
typedef _Float16 v8h  __attribute__((ext_vector_type(8)));
typedef float    v8f  __attribute__((ext_vector_type(8)));

#define BB 512
#define NN 197
#define CC 384
#define LAMW 0.5f

__device__ __forceinline__ float gelu_exact(float x){
    return 0.5f * x * (1.0f + erff(x * 0.70710678118654752f));
}

__device__ __forceinline__ float wave_sum(float v){
#pragma unroll
    for (int off = 16; off > 0; off >>= 1) v += __shfl_xor(v, off, 32);
    return v;
}

// A fragment (16x32 f16) for v_wmma_f32_16x16x32_f16 from row-major LDS tile.
// lane l: M = l&15 ; halfs i<8 -> K = k0 + 8*(l>>4) + i ; i>=8 -> +16.
__device__ __forceinline__ v16h a_frag_lds(const _Float16* Xs, int ldx, int k0){
    int lane = threadIdx.x & 31;
    int M = lane & 15;
    int kk = k0 + ((lane >> 4) << 3);
    const _Float16* p = Xs + M * ldx + kk;
    v8h lo = *(const v8h*)p;
    v8h hi = *(const v8h*)(p + 16);
    v16h r;
#pragma unroll
    for (int i = 0; i < 8; ++i){ r[i] = lo[i]; r[i+8] = hi[i]; }
    return r;
}

// B fragment (32x16 f16) from row-major K x N matrix: lane l = N column n0+(l&15),
// halfs i<8 -> K = k0 + 8*(l>>4) + i ; i>=8 -> +16 (same as A fragment).
__device__ __forceinline__ v16h b_frag_g(const _Float16* W, int ldb, int k0, int n0){
    int lane = threadIdx.x & 31;
    const _Float16* p = W + (size_t)(k0 + ((lane >> 4) << 3)) * ldb + n0 + (lane & 15);
    v16h r;
#pragma unroll
    for (int i = 0; i < 8; ++i){ r[i] = p[(size_t)i * ldb]; r[i+8] = p[(size_t)(i + 16) * ldb]; }
    return r;
}

__device__ __forceinline__ v8f wmma16(v16h a, v16h b, v8f c){
    return __builtin_amdgcn_wmma_f32_16x16x32_f16(false, a, false, b, (short)0, c, false, false);
}

// ---------------------------------------------------------------------------
// Kernel A: fc1_res for all four tensors. grid (512, 4), block 384 (12 waves).
// ---------------------------------------------------------------------------
__global__ void __launch_bounds__(384)
k_fc1res(const float* __restrict__ xq,  const float* __restrict__ xs,
         const float* __restrict__ xqs, const float* __restrict__ xss,
         const float* __restrict__ w1,  const float* __restrict__ b1,
         const float* __restrict__ w2,  const float* __restrict__ b2,
         const float* __restrict__ g,   const float* __restrict__ be,
         _Float16* __restrict__ Hq,  _Float16* __restrict__ Hs,
         _Float16* __restrict__ Hqs, _Float16* __restrict__ Hss,
         float* __restrict__ qc, float* __restrict__ sc,
         float* __restrict__ oqc, float* __restrict__ osc)
{
    int b = blockIdx.x;
    int t = blockIdx.y;
    const float* x; _Float16* Himg; float* cls; float* ocls;
    if (t == 0){ x = xq;  Himg = Hq;  cls = qc; ocls = oqc; }
    else if (t == 1){ x = xs;  Himg = Hs;  cls = sc; ocls = osc; }
    else if (t == 2){ x = xqs; Himg = Hqs; cls = nullptr; ocls = nullptr; }
    else            { x = xss; Himg = Hss; cls = nullptr; ocls = nullptr; }
    x += (size_t)b * NN * CC;

    __shared__ float m[CC];
    __shared__ float h[96];
    __shared__ float red[24];
    int c = threadIdx.x;

    float s = 0.f;
    for (int n = 0; n < NN; ++n) s += x[(size_t)n * CC + c];
    m[c] = s * (1.0f / (float)NN);
    __syncthreads();

    if (c < 96){
        float a = b1[c];
        for (int k = 0; k < CC; ++k) a += m[k] * w1[k * 96 + c];
        h[c] = gelu_exact(a);
    }
    __syncthreads();

    float y = b2[c];
    for (int j = 0; j < 96; ++j) y += h[j] * w2[j * CC + c];

    float gc = g[c], bc = be[c];
    int wid = threadIdx.x >> 5;
    for (int n = 0; n < NN; ++n){
        float v = x[(size_t)n * CC + c] + y;
        float a0 = wave_sum(v);
        float a1 = wave_sum(v * v);
        if ((threadIdx.x & 31) == 0){ red[wid*2] = a0; red[wid*2+1] = a1; }
        __syncthreads();
        float sa = 0.f, sb = 0.f;
        for (int i = 0; i < 12; ++i){ sa += red[i*2]; sb += red[i*2+1]; }
        __syncthreads();
        float mean = sa * (1.0f / (float)CC);
        float var  = sb * (1.0f / (float)CC) - mean * mean;
        float o = (v - mean) * rsqrtf(var + 1e-5f) * gc + bc;
        if (n == 0){
            if (cls){ cls[(size_t)b*CC + c] = o; ocls[(size_t)b*CC + c] = o; }
        } else {
            Himg[((size_t)b*196 + (n-1))*CC + c] = (_Float16)o;
        }
    }
}

// ---------------------------------------------------------------------------
// Weight f32 -> f16 with zero padding
// ---------------------------------------------------------------------------
__global__ void k_cvt_pad(const float* __restrict__ src, _Float16* __restrict__ dst,
                          int sR, int sC, int dR, int dC)
{
    int idx = blockIdx.x * blockDim.x + threadIdx.x;
    if (idx >= dR * dC) return;
    int r = idx / dC, c = idx % dC;
    float v = (r < sR && c < sC) ? src[r * sC + c] : 0.0f;
    dst[idx] = (_Float16)v;
}

__global__ void k_zero16(_Float16* __restrict__ p, size_t n){
    size_t i = (size_t)blockIdx.x * blockDim.x + threadIdx.x;
    if (i < n) p[i] = (_Float16)0.f;
}

// ---------------------------------------------------------------------------
// Kernel B: fc3_path. Rows = (b, channel c), length 196 over tokens.
// grid (12288, 2), block 256 (8 waves). WMMA, K padded 196->224, N 196->208.
// ---------------------------------------------------------------------------
__global__ void __launch_bounds__(256)
k_fc3(_Float16* __restrict__ Hqs, _Float16* __restrict__ Hss,
      const _Float16* __restrict__ W1p, const float* __restrict__ b1,
      const _Float16* __restrict__ W2p, const float* __restrict__ b2,
      const float* __restrict__ g, const float* __restrict__ be)
{
    _Float16* buf = blockIdx.y ? Hss : Hqs;
    int r0 = blockIdx.x * 16;        // row = b*384 + c ; 16 rows share one b
    int bidx = r0 / CC;
    int c0 = r0 % CC;
    _Float16* base = buf + (size_t)bidx * 196 * CC;

    __shared__ _Float16 __align__(16) Xs[16*224];
    __shared__ _Float16 __align__(16) Hl[16*128];
    __shared__ float Ys[16*208];
    __shared__ float meanA[16], rstdA[16];

    int tid = threadIdx.x;
    int rs = tid & 15;
    int nb = tid >> 4;
    for (int it = 0; it < 13; ++it){
        int n = nb + it * 16;                       // 0..207
        float v = (n < 196) ? (float)base[(size_t)n*CC + c0 + rs] : 0.f;
        Xs[rs*224 + n] = (_Float16)v;
    }
    Xs[rs*224 + 208 + nb] = (_Float16)0.f;          // pad 208..223
    __syncthreads();

    int w = tid >> 5, lane = tid & 31;
    {   // GEMM1: (16x224)x(224x128)
        v8f acc = {};
#pragma unroll
        for (int ks = 0; ks < 7; ++ks){
            v16h a = a_frag_lds(Xs, 224, ks*32);
            v16h bf = b_frag_g(W1p, 128, ks*32, w*16);
            acc = wmma16(a, bf, acc);
        }
        int n = w*16 + (lane & 15);
        float bias = b1[n];
        int mo = (lane >> 4) << 3;
#pragma unroll
        for (int j = 0; j < 8; ++j)
            Hl[(j+mo)*128 + n] = (_Float16)gelu_exact(acc[j] + bias);
    }
    __syncthreads();
    for (int t = w; t < 13; t += 8){                // GEMM2: (16x128)x(128x208)
        v8f acc = {};
#pragma unroll
        for (int ks = 0; ks < 4; ++ks){
            v16h a = a_frag_lds(Hl, 128, ks*32);
            v16h bf = b_frag_g(W2p, 208, ks*32, t*16);
            acc = wmma16(a, bf, acc);
        }
        int n = t*16 + (lane & 15);
        float bias = (n < 196) ? b2[n] : 0.f;
        int mo = (lane >> 4) << 3;
#pragma unroll
        for (int j = 0; j < 8; ++j)
            Ys[(j+mo)*208 + n] = acc[j] + bias;
    }
    __syncthreads();
    for (int rr = w*2; rr < w*2 + 2; ++rr){         // LN over 196
        float s = 0.f, s2 = 0.f;
        for (int kc = 0; kc < 7; ++kc){
            int n = lane + kc*32;
            if (n < 196){ float yv = Ys[rr*208 + n]; s += yv; s2 += yv*yv; }
        }
        s = wave_sum(s); s2 = wave_sum(s2);
        if (lane == 0){
            float mean = s * (1.0f/196.0f);
            float var  = s2 * (1.0f/196.0f) - mean*mean;
            meanA[rr] = mean; rstdA[rr] = rsqrtf(var + 1e-5f);
        }
    }
    __syncthreads();
    for (int it = 0; it < 13; ++it){                // coalesced write-back (in place)
        int n = nb + it * 16;
        if (n < 196){
            float yv = Ys[rs*208 + n];
            float o = (yv - meanA[rs]) * rstdA[rs] * g[n] + be[n];
            base[(size_t)n*CC + c0 + rs] = (_Float16)o;
        }
    }
}

// ---------------------------------------------------------------------------
// Kernel C: fc4 + LN + class-combine + l2norm + center. In place over Hq/Hs.
// grid (6272, 2), block 256.
// ---------------------------------------------------------------------------
__global__ void __launch_bounds__(256)
k_fc4(_Float16* __restrict__ Hq, const _Float16* __restrict__ Hqs,
      _Float16* __restrict__ Hs, const _Float16* __restrict__ Hss,
      const _Float16* __restrict__ W1h, const float* __restrict__ b1,
      const _Float16* __restrict__ W2h, const float* __restrict__ b2,
      const float* __restrict__ g, const float* __restrict__ be,
      const float* __restrict__ qc, const float* __restrict__ sc)
{
    _Float16* X = blockIdx.y ? Hs : Hq;
    const _Float16* Xm = blockIdx.y ? Hss : Hqs;
    const float* cls = blockIdx.y ? sc : qc;
    size_t r0 = (size_t)blockIdx.x * 16;

    __shared__ _Float16 __align__(16) Xs[16*384];
    __shared__ _Float16 __align__(16) Hl[16*96];
    __shared__ float Ys[16*384];

    int tid = threadIdx.x;
    for (int it = 0; it < 24; ++it){
        int idx = tid + it*256;
        size_t flat = r0*CC + idx;
        Xs[idx] = (_Float16)((float)X[flat] * (float)Xm[flat]);
    }
    __syncthreads();

    int w = tid >> 5, lane = tid & 31;
    if (w < 6){                                     // GEMM1: (16x384)x(384x96)
        v8f acc = {};
#pragma unroll
        for (int ks = 0; ks < 12; ++ks){
            v16h a = a_frag_lds(Xs, 384, ks*32);
            v16h bf = b_frag_g(W1h, 96, ks*32, w*16);
            acc = wmma16(a, bf, acc);
        }
        int n = w*16 + (lane & 15);
        float bias = b1[n];
        int mo = (lane >> 4) << 3;
#pragma unroll
        for (int j = 0; j < 8; ++j)
            Hl[(j+mo)*96 + n] = (_Float16)gelu_exact(acc[j] + bias);
    }
    __syncthreads();
    for (int t = w; t < 24; t += 8){                // GEMM2: (16x96)x(96x384)
        v8f acc = {};
#pragma unroll
        for (int ks = 0; ks < 3; ++ks){
            v16h a = a_frag_lds(Hl, 96, ks*32);
            v16h bf = b_frag_g(W2h, 384, ks*32, t*16);
            acc = wmma16(a, bf, acc);
        }
        int n = t*16 + (lane & 15);
        float bias = b2[n];
        int mo = (lane >> 4) << 3;
#pragma unroll
        for (int j = 0; j < 8; ++j)
            Ys[(j+mo)*384 + n] = acc[j] + bias;
    }
    __syncthreads();
    for (int rr = w*2; rr < w*2 + 2; ++rr){
        size_t r = r0 + rr;
        int bidx = (int)(r / 196);
        const float* cl = cls + (size_t)bidx * CC;
        float s = 0.f, s2 = 0.f;
#pragma unroll
        for (int kc = 0; kc < 12; ++kc){
            float yv = Ys[rr*384 + lane + kc*32];
            s += yv; s2 += yv*yv;
        }
        s = wave_sum(s); s2 = wave_sum(s2);
        float mean = s * (1.f/384.f);
        float rstd = rsqrtf(s2 * (1.f/384.f) - mean*mean + 1e-5f);
        float z[12]; float sz = 0.f, sz2 = 0.f;
#pragma unroll
        for (int kc = 0; kc < 12; ++kc){
            int n = lane + kc*32;
            float zz = (Ys[rr*384 + n] - mean)*rstd*g[n] + be[n] + LAMW*cl[n];
            z[kc] = zz; sz += zz; sz2 += zz*zz;
        }
        sz = wave_sum(sz); sz2 = wave_sum(sz2);
        float inv = 1.0f / fmaxf(sqrtf(sz2), 1e-12f);
        float mz = sz * (1.f/384.f);
#pragma unroll
        for (int kc = 0; kc < 12; ++kc){
            int n = lane + kc*32;
            X[r*CC + n] = (_Float16)(inv * (z[kc] - mz));
        }
    }
}

// ---------------------------------------------------------------------------
// Kernel T: transpose fs2 (b,196,384) -> FsT (b,384,208) f16, zero padded cols.
// grid (12, 512), block 256.
// ---------------------------------------------------------------------------
__global__ void __launch_bounds__(256)
k_transpose(const _Float16* __restrict__ Hs, _Float16* __restrict__ FsT)
{
    int b = blockIdx.y;
    int kt = blockIdx.x;
    __shared__ _Float16 tile[32*40];
    int tid = threadIdx.x;
    const _Float16* src = Hs + (size_t)b*196*CC;
    _Float16* dst = FsT + (size_t)b*CC*208;
    int k_sub = tid & 31;
    int s_sub = tid >> 5;
    for (int st = 0; st < 7; ++st){
        int s0 = st * 32;
        for (int p = 0; p < 4; ++p){
            int s = s0 + s_sub + p*8;
            float v = (s < 196) ? (float)src[(size_t)s*CC + kt*32 + k_sub] : 0.f;
            tile[(s_sub + p*8)*40 + k_sub] = (_Float16)v;
        }
        __syncthreads();
        int s_out = tid & 31;
        int k_row = tid >> 5;
        for (int p = 0; p < 4; ++p){
            int kr = k_row + p*8;
            int s = s0 + s_out;
            if (s < 208)
                dst[(size_t)(kt*32 + kr)*208 + s] = tile[s_out*40 + kr];
        }
        __syncthreads();
    }
}

// ---------------------------------------------------------------------------
// Kernel D: gram (Fq @ Fs^T) + fused 2x2 pool + square -> Psq f16 (b, 9632).
// grid (13, 512), block 256. Pool done in-registers with shfl_xor.
// ---------------------------------------------------------------------------
__global__ void __launch_bounds__(256)
k_gram(const _Float16* __restrict__ Hq, const _Float16* __restrict__ FsT,
       _Float16* __restrict__ Psq)
{
    int b = blockIdx.y;
    int qt = blockIdx.x;
    const _Float16* A = Hq + (size_t)b*196*CC;
    const _Float16* Bm = FsT + (size_t)b*CC*208;
    __shared__ _Float16 __align__(16) Xs[16*384];

    int tid = threadIdx.x;
    for (int it = 0; it < 24; ++it){
        int idx = tid + it*256;
        int m = idx / 384, k = idx % 384;
        int q = qt*16 + m;
        Xs[idx] = (q < 196) ? A[(size_t)q*CC + k] : (_Float16)0.f;
    }
    __syncthreads();
    int w = tid >> 5, lane = tid & 31;
    for (int st = w; st < 13; st += 8){
        v8f acc = {};
#pragma unroll
        for (int ks = 0; ks < 12; ++ks){
            v16h a = a_frag_lds(Xs, 384, ks*32);
            v16h bf = b_frag_g(Bm, 208, ks*32, st*16);
            acc = wmma16(a, bf, acc);
        }
        // 2x2 pool: vertical = adjacent C vgprs (rows 2i,2i+1), horizontal = lane^1
        float pv[4];
#pragma unroll
        for (int i2 = 0; i2 < 4; ++i2) pv[i2] = acc[2*i2] + acc[2*i2+1];
#pragma unroll
        for (int i2 = 0; i2 < 4; ++i2) pv[i2] += __shfl_xor(pv[i2], 1, 32);
        if ((lane & 1) == 0){
            int pj = st*8 + ((lane & 15) >> 1);
#pragma unroll
            for (int i2 = 0; i2 < 4; ++i2){
                int pi = qt*8 + i2 + 4*(lane >> 4);
                if (pi < 98 && pj < 98){
                    float v = pv[i2] * 0.25f;
                    Psq[(size_t)b*9632 + pi*98 + pj] = (_Float16)(v*v);
                }
            }
        }
    }
}

// ---------------------------------------------------------------------------
// Kernel E: fc2 MLP (9632 -> 256 gelu -> 1). grid 32, block 256.
// ---------------------------------------------------------------------------
__global__ void __launch_bounds__(256)
k_fc2(const _Float16* __restrict__ Psq, const _Float16* __restrict__ W1h,
      const float* __restrict__ b1, const float* __restrict__ w2,
      const float* __restrict__ b2, float* __restrict__ out)
{
    int b0 = blockIdx.x * 16;
    __shared__ _Float16 __align__(16) As[16*32];
    __shared__ float Hl[16*256];
    int tid = threadIdx.x;
    int w = tid >> 5, lane = tid & 31;
    int lm = tid >> 4;
    int lk = (tid & 15) * 2;
    v8f acc0 = {}, acc1 = {};
    for (int ks = 0; ks < 301; ++ks){
        const _Float16* src = Psq + (size_t)(b0 + lm)*9632 + ks*32 + lk;
        As[lm*32 + lk]     = src[0];
        As[lm*32 + lk + 1] = src[1];
        __syncthreads();
        v16h a = a_frag_lds(As, 32, 0);
        v16h bf0 = b_frag_g(W1h, 256, ks*32, w*16);
        v16h bf1 = b_frag_g(W1h, 256, ks*32, (w+8)*16);
        acc0 = wmma16(a, bf0, acc0);
        acc1 = wmma16(a, bf1, acc1);
        __syncthreads();
    }
    int mo = (lane >> 4) << 3;
    int n0 = w*16 + (lane & 15);
    int n1 = (w+8)*16 + (lane & 15);
#pragma unroll
    for (int j = 0; j < 8; ++j){
        Hl[(j+mo)*256 + n0] = gelu_exact(acc0[j] + b1[n0]);
        Hl[(j+mo)*256 + n1] = gelu_exact(acc1[j] + b1[n1]);
    }
    __syncthreads();
    for (int rr = w*2; rr < w*2 + 2; ++rr){
        float p = 0.f;
#pragma unroll
        for (int kc = 0; kc < 8; ++kc){
            int n = lane + kc*32;
            p += Hl[rr*256 + n] * w2[n];
        }
        p = wave_sum(p);
        if (lane == 0) out[b0 + rr] = p + b2[0];
    }
}

// ---------------------------------------------------------------------------
// Kernel F: cs MLP + cosine. grid 512, block 384.
// ---------------------------------------------------------------------------
__global__ void __launch_bounds__(384)
k_cos(const float* __restrict__ qc, const float* __restrict__ sc,
      const float* __restrict__ w1, const float* __restrict__ b1,
      const float* __restrict__ w2, const float* __restrict__ b2,
      float* __restrict__ outc)
{
    int b = blockIdx.x;
    __shared__ float qv[CC], sv[CC], hq[96], hs[96], red[36];
    int c = threadIdx.x;
    qv[c] = qc[(size_t)b*CC + c];
    sv[c] = sc[(size_t)b*CC + c];
    __syncthreads();
    if (c < 96){
        float aq = b1[c], as = b1[c];
        for (int k = 0; k < CC; ++k){
            float wv = w1[k*96 + c];
            aq += qv[k]*wv; as += sv[k]*wv;
        }
        hq[c] = gelu_exact(aq); hs[c] = gelu_exact(as);
    }
    __syncthreads();
    float qx = b2[c], sy = b2[c];
    for (int j = 0; j < 96; ++j){
        float wv = w2[j*CC + c];
        qx += hq[j]*wv; sy += hs[j]*wv;
    }
    float d = wave_sum(qx*sy), nq = wave_sum(qx*qx), ns = wave_sum(sy*sy);
    int wid = threadIdx.x >> 5;
    if ((threadIdx.x & 31) == 0){ red[wid*3]=d; red[wid*3+1]=nq; red[wid*3+2]=ns; }
    __syncthreads();
    if (threadIdx.x == 0){
        float sd=0.f, sq=0.f, ss=0.f;
        for (int i = 0; i < 12; ++i){ sd+=red[i*3]; sq+=red[i*3+1]; ss+=red[i*3+2]; }
        outc[b] = sd / (fmaxf(sqrtf(sq),1e-8f) * fmaxf(sqrtf(ss),1e-8f));
    }
}

// ---------------------------------------------------------------------------
extern "C" void kernel_launch(void* const* d_in, const int* in_sizes, int n_in,
                              void* d_out, int out_size, void* d_ws, size_t ws_size,
                              hipStream_t stream)
{
    const float* f_q    = (const float*)d_in[0];
    const float* f_s    = (const float*)d_in[1];
    const float* f_qs   = (const float*)d_in[2];
    const float* f_ss   = (const float*)d_in[3];
    const float* fc1_w1 = (const float*)d_in[4];
    const float* fc1_b1 = (const float*)d_in[5];
    const float* fc1_w2 = (const float*)d_in[6];
    const float* fc1_b2 = (const float*)d_in[7];
    const float* ln1_g  = (const float*)d_in[8];
    const float* ln1_b  = (const float*)d_in[9];
    const float* fc3_w1 = (const float*)d_in[10];
    const float* fc3_b1 = (const float*)d_in[11];
    const float* fc3_w2 = (const float*)d_in[12];
    const float* fc3_b2 = (const float*)d_in[13];
    const float* ln3_g  = (const float*)d_in[14];
    const float* ln3_b  = (const float*)d_in[15];
    const float* fc4_w1 = (const float*)d_in[16];
    const float* fc4_b1 = (const float*)d_in[17];
    const float* fc4_w2 = (const float*)d_in[18];
    const float* fc4_b2 = (const float*)d_in[19];
    const float* ln4_g  = (const float*)d_in[20];
    const float* ln4_b  = (const float*)d_in[21];
    const float* fc2_w1 = (const float*)d_in[22];
    const float* fc2_b1 = (const float*)d_in[23];
    const float* fc2_w2 = (const float*)d_in[24];
    const float* fc2_b2 = (const float*)d_in[25];
    const float* cs_w1  = (const float*)d_in[26];
    const float* cs_b1  = (const float*)d_in[27];
    const float* cs_w2  = (const float*)d_in[28];
    const float* cs_b2  = (const float*)d_in[29];

    char* ws = (char*)d_ws;
    size_t off = 0;
    auto alloc = [&](size_t bytes) -> void* {
        void* p = ws + off;
        off += (bytes + 255) & ~(size_t)255;
        return p;
    };
    const size_t SZH = (size_t)BB*196*CC*sizeof(_Float16);
    _Float16* Hq  = (_Float16*)alloc(SZH);
    _Float16* Hs  = (_Float16*)alloc(SZH);
    _Float16* Hqs = (_Float16*)alloc(SZH);
    _Float16* Hss = (_Float16*)alloc(SZH);
    _Float16* FsT = (_Float16*)alloc((size_t)BB*CC*208*sizeof(_Float16));
    float* qc = (float*)alloc((size_t)BB*CC*sizeof(float));
    float* sc = (float*)alloc((size_t)BB*CC*sizeof(float));
    _Float16* Psq = (_Float16*)alloc((size_t)BB*9632*sizeof(_Float16));
    _Float16* W3a = (_Float16*)alloc((size_t)224*128*sizeof(_Float16));
    _Float16* W3b = (_Float16*)alloc((size_t)128*208*sizeof(_Float16));
    _Float16* W4a = (_Float16*)alloc((size_t)384*96*sizeof(_Float16));
    _Float16* W4b = (_Float16*)alloc((size_t)96*384*sizeof(_Float16));
    _Float16* W2a = (_Float16*)alloc((size_t)9632*256*sizeof(_Float16));

    float* out      = (float*)d_out;
    float* out_cos  = out + BB;
    float* out_qcls = out + 2*BB;
    float* out_scls = out + 2*BB + (size_t)BB*CC;

    auto cvt = [&](const float* s, _Float16* dd, int sR, int sC, int dR, int dC){
        int total = dR * dC;
        k_cvt_pad<<<(total + 255)/256, 256, 0, stream>>>(s, dd, sR, sC, dR, dC);
    };
    cvt(fc3_w1, W3a, 196, 128, 224, 128);
    cvt(fc3_w2, W3b, 128, 196, 128, 208);
    cvt(fc4_w1, W4a, 384, 96, 384, 96);
    cvt(fc4_w2, W4b, 96, 384, 96, 384);
    cvt(fc2_w1, W2a, 9604, 256, 9632, 256);
    {
        size_t n = (size_t)BB*9632;
        k_zero16<<<(unsigned)((n + 255)/256), 256, 0, stream>>>(Psq, n);
    }

    k_fc1res<<<dim3(BB, 4), 384, 0, stream>>>(
        f_q, f_s, f_qs, f_ss, fc1_w1, fc1_b1, fc1_w2, fc1_b2, ln1_g, ln1_b,
        Hq, Hs, Hqs, Hss, qc, sc, out_qcls, out_scls);

    k_fc3<<<dim3((BB*CC)/16, 2), 256, 0, stream>>>(
        Hqs, Hss, W3a, fc3_b1, W3b, fc3_b2, ln3_g, ln3_b);

    k_fc4<<<dim3((BB*196)/16, 2), 256, 0, stream>>>(
        Hq, Hqs, Hs, Hss, W4a, fc4_b1, W4b, fc4_b2, ln4_g, ln4_b, qc, sc);

    k_transpose<<<dim3(12, BB), 256, 0, stream>>>(Hs, FsT);

    k_gram<<<dim3(13, BB), 256, 0, stream>>>(Hq, FsT, Psq);

    k_fc2<<<BB/16, 256, 0, stream>>>(Psq, W2a, fc2_b1, fc2_w2, fc2_b2, out);

    k_cos<<<BB, 384, 0, stream>>>(qc, sc, cs_w1, cs_b1, cs_w2, cs_b2, out_cos);
}
